// PIKANBlock_75007308857406
// MI455X (gfx1250) — matrix operand order repaired and checked by reading, not compiled, not run on hardware
//
#include <hip/hip_runtime.h>

typedef __attribute__((ext_vector_type(8)))  float    v8f;
typedef __attribute__((ext_vector_type(8)))  _Float16 v8h;
typedef __attribute__((ext_vector_type(16))) _Float16 v16h;

#define B_PTS     2048
#define GRID_SZ   253
#define KTOT      256          // GRID_SIZE + SPLINE_ORDER (padded per-i stride)
#define NOUT      256
#define MT        64           // M rows per workgroup
#define LDSP      264          // basis row pitch in f16 (256 + 8 pad, keeps 16B align)

#define SHUF16(lo, hi) __builtin_shufflevector(lo, hi, \
    0, 1, 2, 3, 4, 5, 6, 7, 8, 9, 10, 11, 12, 13, 14, 15)

// IGroupLP scheduling masks
#define SCHED_MFMA      0x008   // MFMA / WMMA
#define SCHED_VMEM_READ 0x020
#define SCHED_DS_READ   0x100

// ---------------------------------------------------------------------------
// Pre-pass: transpose coeffs [I*256, 256] fp32 -> [256, I*256] f16, folding the
// base weights into padding slot g==253 and zeroing g==254/255.
// ---------------------------------------------------------------------------
__global__ __launch_bounds__(256)
void kan_transpose_pack(const float* __restrict__ src,   // [R, 256]
                        const float* __restrict__ bw,    // [I, 256]
                        _Float16* __restrict__ dst,      // [256, R] f16
                        int R) {
  __shared__ float tile[32][33];
  const int r0 = blockIdx.x * 32;
  const int k0 = blockIdx.y * 32;
  const int tx = threadIdx.x;          // 0..31
  const int ty = threadIdx.y;          // 0..7
#pragma unroll
  for (int j = 0; j < 4; ++j) {
    const int r = ty + j * 8;
    tile[r][tx] = src[(size_t)(r0 + r) * 256 + (k0 + tx)];
  }
  __syncthreads();
#pragma unroll
  for (int j = 0; j < 4; ++j) {
    const int k  = k0 + ty + j * 8;    // output row (spline-K axis)
    const int rr = r0 + tx;            // output col (flattened i*256+g)
    float v = tile[tx][ty + j * 8];
    const int g = rr & 255;
    const int i = rr >> 8;
    if (g == 253)      v = bw[(size_t)i * 256 + k];   // fold base term
    else if (g > 253)  v = 0.0f;                      // torch zero-padding
    dst[(size_t)k * R + rr] = (_Float16)v;
  }
}

// ---------------------------------------------------------------------------
// Fused KAN layer: Y[b,k] = sum_{i,g} basis(X[b,i],g) * T[k, i*256+g]
// (base term folded into slot g==253). Windowed basis: only ~16 grid slots
// per (row,i) are non-zero in f16 (lossless vs dense f16), so fill/clear just
// that window in a persistently-zeroed LDS tile.
// WG tile: M=64 x N=64. 8 waves in 2x4 grid of M32xN16 register tiles.
// Steady-state schedule pinned via sched_group_barrier:
//   [DS4 (A c+1)] [VMEM2 (B c+1)] [WMMA2 (c)]
// ---------------------------------------------------------------------------
__global__ __launch_bounds__(256)
void kan_layer_wmma(const float* __restrict__ X,      // [B, I]
                    int I,
                    const _Float16* __restrict__ T,   // [256, I*256] f16
                    const float* __restrict__ grid,   // [253]
                    float* __restrict__ Y) {          // [B, 256]
  __shared__ __align__(32) _Float16 basis[MT * LDSP]; // 64 rows x 256 f16 (+pad)
  __shared__ float sgrid[256];

  const int tid = threadIdx.x;
  sgrid[tid] = (tid < GRID_SZ) ? grid[tid] : 0.0f;
  {
    unsigned int* bz = (unsigned int*)basis;
    const int nwords = (MT * LDSP) / 2;               // 8448
#pragma unroll 4
    for (int z = tid; z < nwords; z += 256) bz[z] = 0u;
  }

  const int lane = tid & 31;
  const int wave = tid >> 5;
  const int b0   = blockIdx.x * MT;     // M-tile base row
  const int n0   = blockIdx.y * 64;     // N-split base column
  const int mh   = (wave >> 2) * 32;    // wave's M sub-tile (0 or 32)
  const int col  = n0 + (wave & 3) * 16;// wave's 16 output columns
  const size_t LDK = (size_t)I * KTOT;  // contraction length (padded)

  // Fragment lane mapping (16-bit A / B: contiguous K order, lane half h holds k = 16h + s)
  const int nlane = lane & 15;          // B/C/D column within tile
  const int khalf = lane >> 4;          // 0: K-low half, 1: K-high half
  const int arow  = lane & 15;          // A row within 16-row tile
  const int kb    = khalf * 16;         // A K sub-offset within chunk

  // Basis producer mapping: 4 consecutive lanes (same wave!) per row, so the
  // clear-then-fill DS stores for one row are issued in order by one wave.
  const int prow  = tid >> 2;           // 0..63
  const int pslot = (tid & 3) * 4;      // 0,4,8,12 -> 4 slots each

  // -0.5 * log2(e) / sigma^2, sigma = 2/253
  const float KC = -0.5f * 1.44269504088896340736f * (253.0f * 253.0f * 0.25f);

  const _Float16* bpw = T + (size_t)(col + nlane) * LDK + khalf * 16;
  const int a0off = (mh + arow) * LDSP + kb;
  const int a1off = a0off + 16 * LDSP;

  v8f acc0 = {};
  v8f acc1 = {};
  int w0prev = 0;

  __syncthreads();                      // sgrid + zeroed basis ready

  for (int i = 0; i < I; ++i) {
    // ---- clear old window, fill new window ----
    const float x  = X[(size_t)(b0 + prow) * I + i];
    const float xn = fminf(fmaxf(x, -1.0f), 1.0f);
    const float gx = (xn + 1.0f) * 126.0f;            // continuous grid index
    int w0 = (int)gx - 7;
    w0 = (w0 < 0) ? 0 : ((w0 > 240) ? 240 : w0);
    _Float16* brow = &basis[prow * LDSP];
#pragma unroll
    for (int s = 0; s < 4; ++s) {                     // clear previous window
      const int g = w0prev + pslot + s;
      if (g < GRID_SZ) brow[g] = (_Float16)0.0f;
    }
    if ((tid & 3) == 0) brow[GRID_SZ] = (_Float16)0.0f;
#pragma unroll
    for (int s = 0; s < 4; ++s) {                     // fill current window
      const int g = w0 + pslot + s;
      if (g < GRID_SZ) {
        const float d = xn - sgrid[g];
        brow[g] = (_Float16)exp2f(KC * d * d);
      }
    }
    if ((tid & 3) == 0) {                             // silu(x) at folded slot
      brow[GRID_SZ] = (_Float16)(x / (1.0f + exp2f(-1.44269504088896340736f * x)));
    }
    w0prev = w0;
    __syncthreads();

    // ---- 8 chunks of K=32, software-pipelined one chunk ahead ----
    const _Float16* bpi = bpw + (size_t)i * KTOT;
    v16h a0c, a1c, bfc;
    {
      const v8h l0 = *(const v8h*)&basis[a0off];
      const v8h h0 = *(const v8h*)&basis[a0off + 8];
      const v8h l1 = *(const v8h*)&basis[a1off];
      const v8h h1 = *(const v8h*)&basis[a1off + 8];
      a0c = SHUF16(l0, h0);
      a1c = SHUF16(l1, h1);
      bfc = *(const v16h*)bpi;
    }
    // prologue pipeline stage: 4 DS reads then 2 VMEM reads
    __builtin_amdgcn_sched_group_barrier(SCHED_DS_READ,   4, 0);
    __builtin_amdgcn_sched_group_barrier(SCHED_VMEM_READ, 2, 0);
#pragma unroll
    for (int c = 0; c < 7; ++c) {
      const int gn = (c + 1) * 32;
      // prefetch next chunk's fragments into fresh registers
      const v8h l0 = *(const v8h*)&basis[a0off + gn];
      const v8h h0 = *(const v8h*)&basis[a0off + gn + 8];
      const v8h l1 = *(const v8h*)&basis[a1off + gn];
      const v8h h1 = *(const v8h*)&basis[a1off + gn + 8];
      const v16h bfn = *(const v16h*)(bpi + gn);
      // consume current chunk
      acc0 = __builtin_amdgcn_wmma_f32_16x16x32_f16(
          false, a0c, false, bfc, (short)0, acc0, false, false);
      acc1 = __builtin_amdgcn_wmma_f32_16x16x32_f16(
          false, a1c, false, bfc, (short)0, acc1, false, false);
      a0c = SHUF16(l0, h0);
      a1c = SHUF16(l1, h1);
      bfc = bfn;
      // steady-state pipeline stage: loads of c+1 before WMMAs of c
      __builtin_amdgcn_sched_group_barrier(SCHED_DS_READ,   4, 0);
      __builtin_amdgcn_sched_group_barrier(SCHED_VMEM_READ, 2, 0);
      __builtin_amdgcn_sched_group_barrier(SCHED_MFMA,      2, 0);
    }
    acc0 = __builtin_amdgcn_wmma_f32_16x16x32_f16(
        false, a0c, false, bfc, (short)0, acc0, false, false);
    acc1 = __builtin_amdgcn_wmma_f32_16x16x32_f16(
        false, a1c, false, bfc, (short)0, acc1, false, false);
    __builtin_amdgcn_sched_group_barrier(SCHED_MFMA, 2, 0);
    __syncthreads();                    // A-frag reads done before next clear
  }

  // ---- store C/D (lanes 0-15: M=r, lanes 16-31: M=8+r; N = lane&15) ----
#pragma unroll
  for (int r = 0; r < 8; ++r) {
    const int m = b0 + mh + khalf * 8 + r;
    Y[(size_t)m * NOUT + col + nlane]        = acc0[r];
    Y[(size_t)(m + 16) * NOUT + col + nlane] = acc1[r];
  }
}

// ---------------------------------------------------------------------------
extern "C" void kernel_launch(void* const* d_in, const int* in_sizes, int n_in,
                              void* d_out, int out_size, void* d_ws, size_t ws_size,
                              hipStream_t stream) {
  (void)in_sizes; (void)n_in; (void)out_size; (void)ws_size;
  const float* x_in    = (const float*)d_in[0];   // [2048, 4]
  const float* grid    = (const float*)d_in[1];   // [253]
  const float* coeffs1 = (const float*)d_in[2];   // [4, 256, 256]
  const float* bw1     = (const float*)d_in[3];   // [4, 256]
  const float* coeffs2 = (const float*)d_in[4];   // [256, 256, 256]
  const float* bw2     = (const float*)d_in[5];   // [256, 256]
  float* out = (float*)d_out;                     // [2048, 256]

  char* ws = (char*)d_ws;
  const size_t T2_BYTES = (size_t)256 * 65536 * sizeof(_Float16); // 33.5 MB
  const size_t T1_BYTES = (size_t)256 * 1024  * sizeof(_Float16); // 0.5 MB
  _Float16* T2     = (_Float16*)ws;
  _Float16* T1     = (_Float16*)(ws + T2_BYTES);
  float*    hidden = (float*)   (ws + T2_BYTES + T1_BYTES);       // [2048,256]

  const dim3 tb(32, 8);
  // Layer 1 tables: R = 4*256 = 1024 rows
  kan_transpose_pack<<<dim3(1024 / 32, 256 / 32), tb, 0, stream>>>(
      coeffs1, bw1, T1, 1024);
  // Layer 2 tables: R = 256*256 = 65536 rows
  kan_transpose_pack<<<dim3(65536 / 32, 256 / 32), tb, 0, stream>>>(
      coeffs2, bw2, T2, 65536);

  // Layer 1: I = 4, X = inputs -> hidden
  kan_layer_wmma<<<dim3(B_PTS / MT, 4), 256, 0, stream>>>(x_in, 4, T1, grid, hidden);
  // Layer 2: I = 256, X = hidden -> out
  kan_layer_wmma<<<dim3(B_PTS / MT, 4), 256, 0, stream>>>(hidden, 256, T2, grid, out);
}
